// UnifiedGlobalEncoder_88330297409790
// MI455X (gfx1250) — matrix operand order repaired and checked by reading, not compiled, not run on hardware
//
#include <hip/hip_runtime.h>
#include <cstdint>
#include <cstddef>

#define N_NODES 100000
#define HID 256
#define N_REL 3
#define K_TEXT 384
#define K_IMG  512
#define LN_EPS 1e-5f
#define DENOM_EPS 1e-8f

typedef float v8f  __attribute__((ext_vector_type(8)));
typedef __bf16 v16bf __attribute__((ext_vector_type(16)));
typedef unsigned int v8u __attribute__((ext_vector_type(8)));

union BFU { v8u u; v16bf b; };

// round-to-nearest-even f32 -> bf16 bits (used offline for weight packing)
__device__ __forceinline__ unsigned bf16_rne_bits(float x)
{
    unsigned u = __float_as_uint(x);
    return (u + 0x7FFFu + ((u >> 16) & 1u)) >> 16;
}

// pack the truncated-bf16 of two floats into one dword: {bf16(x1), bf16(x0)}
__device__ __forceinline__ unsigned pack_trunc_bf16(float x0, float x1)
{
    return __builtin_amdgcn_perm(__float_as_uint(x1), __float_as_uint(x0), 0x07060302u);
}

// ---------------------------------------------------------------------------
// Load one 16x32 bf16 A-fragment pair (hi/lo split) from a row-major f32 row.
// CDNA5 16-bit A layout: lane<16 holds K {0..7, 16..23}; lane>=16 holds
// K {8..15, 24..31}; VGPR i packs K=2i,2i+1 of its half.
// hi = truncated bf16 of x; lo = truncated bf16 of exact residual x - hi.
// 6 VALU per element pair (2x v_perm, 2x v_and, 2x v_sub).
// ---------------------------------------------------------------------------
__device__ __forceinline__ void load_split_afrag(const float* __restrict__ p, int hf,
                                                 v16bf& hi, v16bf& lo)
{
    const float4* q0 = (const float4*)(p + 8 * hf);
    const float4* q1 = (const float4*)(p + 16 + 8 * hf);
    float4 c0 = q0[0], c1 = q0[1], c2 = q1[0], c3 = q1[1];
    float xs[16] = {c0.x, c0.y, c0.z, c0.w, c1.x, c1.y, c1.z, c1.w,
                    c2.x, c2.y, c2.z, c2.w, c3.x, c3.y, c3.z, c3.w};
    BFU H, L;
#pragma unroll
    for (int i = 0; i < 8; ++i) {
        float x0 = xs[2 * i], x1 = xs[2 * i + 1];
        H.u[i] = pack_trunc_bf16(x0, x1);
        float r0 = x0 - __uint_as_float(__float_as_uint(x0) & 0xFFFF0000u);
        float r1 = x1 - __uint_as_float(__float_as_uint(x1) & 0xFFFF0000u);
        L.u[i] = pack_trunc_bf16(r0, r1);
    }
    hi = H.b;
    lo = L.b;
}

#define WMMA_BF16(A, B, C) \
    __builtin_amdgcn_wmma_f32_16x16x32_bf16(false, (A), false, (B), (short)0, (C), false, false)

// ---------------------------------------------------------------------------
// bf16x3 GEMM pass over K for TWO 16-row M-tiles sharing B fragments.
// a0/a1: per-lane row pointers (A + (m0[+16] + l16) * lda).
// Bhi/Blo: packed weights, layout [K/32][256][32] bf16, so each lane's
// fragment is a contiguous 32B load.  acc layout = CDNA5 16x16 f32 C/D.
// ---------------------------------------------------------------------------
__device__ __forceinline__ void gemm_pass_bf16x3(
    const float* __restrict__ a0, const float* __restrict__ a1, int K,
    const unsigned short* __restrict__ Bhi, const unsigned short* __restrict__ Blo,
    int n0, int hf, int l16, v8f acc0[4], v8f acc1[4])
{
    const int chunks = K >> 5;
    for (int kc = 0; kc < chunks; ++kc) {
        v16bf a0h, a0l, a1h, a1l;
        load_split_afrag(a0 + kc * 32, hf, a0h, a0l);
        load_split_afrag(a1 + kc * 32, hf, a1h, a1l);
        const unsigned short* bh = Bhi + (size_t)kc * (HID * 32) + hf * 16;
        const unsigned short* bl = Blo + (size_t)kc * (HID * 32) + hf * 16;
#pragma unroll
        for (int t = 0; t < 4; ++t) {
            const int n = n0 + t * 16 + l16;
            v16bf bhi = *(const v16bf*)(bh + (size_t)n * 32);
            v16bf blo = *(const v16bf*)(bl + (size_t)n * 32);
            acc0[t] = WMMA_BF16(a0h, bhi, acc0[t]);
            acc0[t] = WMMA_BF16(a0l, bhi, acc0[t]);
            acc0[t] = WMMA_BF16(a0h, blo, acc0[t]);
            acc1[t] = WMMA_BF16(a1h, bhi, acc1[t]);
            acc1[t] = WMMA_BF16(a1l, bhi, acc1[t]);
            acc1[t] = WMMA_BF16(a1h, blo, acc1[t]);
        }
    }
}

// ---------------------------------------------------------------------------
// Pack a row-major f32 weight [K, 256] into hi/lo bf16 fragment layout
// [K/32][256][32].  RNE splits (offline, so take the better rounding).
// ---------------------------------------------------------------------------
__global__ void uge_pack_w_kernel(const float* __restrict__ W, int K,
                                  unsigned short* __restrict__ hi,
                                  unsigned short* __restrict__ lo)
{
    int t = blockIdx.x * blockDim.x + threadIdx.x;
    int total = (K >> 5) * HID;
    if (t >= total) return;
    int chunk = t >> 8;
    int n = t & 255;
    size_t obase = (size_t)chunk * (HID * 32) + (size_t)n * 32;
#pragma unroll 4
    for (int j = 0; j < 32; ++j) {
        float x = W[(size_t)(chunk * 32 + j) * HID + n];
        unsigned hb = bf16_rne_bits(x);
        float r = x - __uint_as_float(hb << 16);
        unsigned lb = bf16_rne_bits(r);
        const int jp = ((j >> 3) & 1) * 16 + (j & 7) + ((j >> 4) << 3);
        hi[obase + jp] = (unsigned short)hb;
        lo[obase + jp] = (unsigned short)lb;
    }
}

// ---------------------------------------------------------------------------
// h0 = 0.5 * (feat_text @ W_text + feat_image @ W_image); h0[0,:] = 0
// grid: N/32 blocks, 128 threads (4 waves; each wave: 32(M) x 64(N)).
// ---------------------------------------------------------------------------
__global__ void __launch_bounds__(128, 1)
uge_embed_kernel(const float* __restrict__ ft, const float* __restrict__ fi,
                 const unsigned short* __restrict__ Wt_hi,
                 const unsigned short* __restrict__ Wt_lo,
                 const unsigned short* __restrict__ Wi_hi,
                 const unsigned short* __restrict__ Wi_lo,
                 float* __restrict__ h0)
{
    const int lane = threadIdx.x & 31;
    const int wv   = threadIdx.x >> 5;
    const int m0   = blockIdx.x * 32;
    const int n0   = wv * 64;
    const int hf   = lane >> 4;
    const int l16  = lane & 15;

    v8f acc0[4] = {}, acc1[4] = {};
    gemm_pass_bf16x3(ft + (size_t)(m0 + l16) * K_TEXT, ft + (size_t)(m0 + 16 + l16) * K_TEXT,
                     K_TEXT, Wt_hi, Wt_lo, n0, hf, l16, acc0, acc1);
    gemm_pass_bf16x3(fi + (size_t)(m0 + l16) * K_IMG, fi + (size_t)(m0 + 16 + l16) * K_IMG,
                     K_IMG, Wi_hi, Wi_lo, n0, hf, l16, acc0, acc1);

#pragma unroll
    for (int t = 0; t < 4; ++t) {
#pragma unroll
        for (int v = 0; v < 8; ++v) {
            int row0 = m0 + v + 8 * hf;
            int row1 = row0 + 16;
            float v0 = 0.5f * acc0[t][v];
            float v1 = 0.5f * acc1[t][v];
            if (row0 == 0) v0 = 0.0f;
            h0[(size_t)row0 * HID + n0 + t * 16 + l16] = v0;
            h0[(size_t)row1 * HID + n0 + t * 16 + l16] = v1;
        }
    }
}

// ---------------------------------------------------------------------------
// out[n,:] += (Aagg[n,:] @ W_rel) / max(denom[n], eps)
// ---------------------------------------------------------------------------
__global__ void __launch_bounds__(128, 1)
uge_relmm_kernel(const float* __restrict__ Aagg,
                 const unsigned short* __restrict__ W_hi,
                 const unsigned short* __restrict__ W_lo,
                 const float* __restrict__ denom, float* __restrict__ out)
{
    const int lane = threadIdx.x & 31;
    const int wv   = threadIdx.x >> 5;
    const int m0   = blockIdx.x * 32;
    const int n0   = wv * 64;
    const int hf   = lane >> 4;
    const int l16  = lane & 15;

    v8f acc0[4] = {}, acc1[4] = {};
    gemm_pass_bf16x3(Aagg + (size_t)(m0 + l16) * HID, Aagg + (size_t)(m0 + 16 + l16) * HID,
                     HID, W_hi, W_lo, n0, hf, l16, acc0, acc1);

#pragma unroll
    for (int t = 0; t < 4; ++t) {
#pragma unroll
        for (int v = 0; v < 8; ++v) {
            int row0 = m0 + v + 8 * hf;
            int row1 = row0 + 16;
            float s0 = 1.0f / fmaxf(denom[row0], DENOM_EPS);
            float s1 = 1.0f / fmaxf(denom[row1], DENOM_EPS);
            out[(size_t)row0 * HID + n0 + t * 16 + l16] += acc0[t][v] * s0;
            out[(size_t)row1 * HID + n0 + t * 16 + l16] += acc1[t][v] * s1;
        }
    }
}

// ---------------------------------------------------------------------------
// denom[rel][dst] += w   (one thread per edge)
// ---------------------------------------------------------------------------
__global__ void uge_denom_kernel(const int* __restrict__ dst, const int* __restrict__ etype,
                                 const float* __restrict__ ew, int E, float* __restrict__ denom)
{
    int e = blockIdx.x * blockDim.x + threadIdx.x;
    if (e >= E) return;
    atomicAdd(denom + (size_t)etype[e] * N_NODES + dst[e], ew[e]);
}

// ---------------------------------------------------------------------------
// Aagg[dst,:] += w * h0[src,:] for edges of relation `rel` (one wave per edge)
// ---------------------------------------------------------------------------
__global__ void uge_scatter_kernel(const float* __restrict__ h0,
                                   const int* __restrict__ src, const int* __restrict__ dst,
                                   const int* __restrict__ etype, const float* __restrict__ ew,
                                   int E, int rel, float* __restrict__ Aagg)
{
    const int wpb = blockDim.x >> 5;
    int e = blockIdx.x * wpb + (threadIdx.x >> 5);
    if (e >= E) return;
    if (etype[e] != rel) return;
    const float w = ew[e];
    const int lane = threadIdx.x & 31;
    const float4* hs = (const float4*)(h0 + (size_t)src[e] * HID);
    float* ad = Aagg + (size_t)dst[e] * HID;
#pragma unroll
    for (int i = 0; i < 2; ++i) {
        float4 v = hs[lane + 32 * i];
        int c = (lane + 32 * i) * 4;
        atomicAdd(ad + c + 0, w * v.x);
        atomicAdd(ad + c + 1, w * v.y);
        atomicAdd(ad + c + 2, w * v.z);
        atomicAdd(ad + c + 3, w * v.w);
    }
}

// ---------------------------------------------------------------------------
// out[n,:] = LayerNorm(h0[n,:] + out[n,:]) * gamma + beta; out[0,:] = 0
// ---------------------------------------------------------------------------
__global__ void uge_ln_kernel(const float* __restrict__ h0, float* __restrict__ out,
                              const float* __restrict__ gamma, const float* __restrict__ beta)
{
    const int wpb = blockDim.x >> 5;
    int row = blockIdx.x * wpb + (threadIdx.x >> 5);
    if (row >= N_NODES) return;
    const int lane = threadIdx.x & 31;

    const float4* h4 = (const float4*)(h0 + (size_t)row * HID);
    float4* o4 = (float4*)(out + (size_t)row * HID);

    float4 xa = h4[lane * 2 + 0];
    float4 xb = h4[lane * 2 + 1];
    float4 aa = o4[lane * 2 + 0];
    float4 ab = o4[lane * 2 + 1];
    xa.x += aa.x; xa.y += aa.y; xa.z += aa.z; xa.w += aa.w;
    xb.x += ab.x; xb.y += ab.y; xb.z += ab.z; xb.w += ab.w;

    float s = xa.x + xa.y + xa.z + xa.w + xb.x + xb.y + xb.z + xb.w;
#pragma unroll
    for (int off = 16; off >= 1; off >>= 1) s += __shfl_xor(s, off, 32);
    const float mu = s * (1.0f / HID);

    float d, var = 0.0f;
    d = xa.x - mu; var += d * d; d = xa.y - mu; var += d * d;
    d = xa.z - mu; var += d * d; d = xa.w - mu; var += d * d;
    d = xb.x - mu; var += d * d; d = xb.y - mu; var += d * d;
    d = xb.z - mu; var += d * d; d = xb.w - mu; var += d * d;
#pragma unroll
    for (int off = 16; off >= 1; off >>= 1) var += __shfl_xor(var, off, 32);
    var *= (1.0f / HID);
    const float r = rsqrtf(var + LN_EPS);

    const float4* g4 = (const float4*)gamma;
    const float4* b4 = (const float4*)beta;
    float4 ga = g4[lane * 2 + 0], gb = g4[lane * 2 + 1];
    float4 ba = b4[lane * 2 + 0], bb = b4[lane * 2 + 1];

    float4 ya, yb;
    ya.x = (xa.x - mu) * r * ga.x + ba.x; ya.y = (xa.y - mu) * r * ga.y + ba.y;
    ya.z = (xa.z - mu) * r * ga.z + ba.z; ya.w = (xa.w - mu) * r * ga.w + ba.w;
    yb.x = (xb.x - mu) * r * gb.x + bb.x; yb.y = (xb.y - mu) * r * gb.y + bb.y;
    yb.z = (xb.z - mu) * r * gb.z + bb.z; yb.w = (xb.w - mu) * r * gb.w + bb.w;
    if (row == 0) { ya = make_float4(0, 0, 0, 0); yb = make_float4(0, 0, 0, 0); }
    o4[lane * 2 + 0] = ya;
    o4[lane * 2 + 1] = yb;
}

__global__ void uge_zero_kernel(float* __restrict__ p, size_t n)
{
    size_t i = (size_t)blockIdx.x * blockDim.x + threadIdx.x;
    size_t stride = (size_t)gridDim.x * blockDim.x;
    for (; i < n; i += stride) p[i] = 0.0f;
}

extern "C" void kernel_launch(void* const* d_in, const int* in_sizes, int n_in,
                              void* d_out, int out_size, void* d_ws, size_t ws_size,
                              hipStream_t stream)
{
    (void)n_in; (void)out_size; (void)ws_size;

    const float* feat_text  = (const float*)d_in[0];   // [N, 384]
    const float* feat_image = (const float*)d_in[1];   // [N, 512]
    const float* W_text     = (const float*)d_in[2];   // [384, 256]
    const float* W_image    = (const float*)d_in[3];   // [512, 256]
    const float* W_rel      = (const float*)d_in[4];   // [3, 256, 256]
    const float* ln_gamma   = (const float*)d_in[5];   // [256]
    const float* ln_beta    = (const float*)d_in[6];   // [256]
    const float* edge_w     = (const float*)d_in[7];   // [E]
    const int*   edge_index = (const int*)d_in[8];     // [2, E]
    const int*   edge_type  = (const int*)d_in[9];     // [E]
    const int E = in_sizes[7];
    const int* src = edge_index;
    const int* dst = edge_index + E;

    const size_t NH = (size_t)N_NODES * HID;
    float* out   = (float*)d_out;            // [N, 256] (doubles as agg accumulator)
    float* h0    = (float*)d_ws;             // [N, 256]
    float* Aagg  = h0 + NH;                  // [N, 256] per-relation scratch
    float* denom = Aagg + NH;                // [3, N]

    // packed bf16 hi/lo weights (fragment-native layout), 32B-aligned region
    unsigned short* pk    = (unsigned short*)(denom + (size_t)N_REL * N_NODES);
    unsigned short* Wt_hi = pk;                       // 384*256
    unsigned short* Wt_lo = Wt_hi + 98304;
    unsigned short* Wi_hi = Wt_lo + 98304;            // 512*256
    unsigned short* Wi_lo = Wi_hi + 131072;
    unsigned short* Wr_hi = Wi_lo + 131072;           // 3*256*256 (rel-major chunks)
    unsigned short* Wr_lo = Wr_hi + 196608;

    const int mblocks = N_NODES / 32;                 // 3125

    // 0) pack weights into bf16 hi/lo fragment layout
    uge_pack_w_kernel<<<(3072 + 255) / 256, 256, 0, stream>>>(W_text, K_TEXT, Wt_hi, Wt_lo);
    uge_pack_w_kernel<<<(4096 + 255) / 256, 256, 0, stream>>>(W_image, K_IMG, Wi_hi, Wi_lo);
    uge_pack_w_kernel<<<(6144 + 255) / 256, 256, 0, stream>>>(W_rel, N_REL * HID, Wr_hi, Wr_lo);

    // 1) zero aggregation accumulator (d_out) and denominators
    uge_zero_kernel<<<2048, 256, 0, stream>>>(out, NH);
    uge_zero_kernel<<<512, 256, 0, stream>>>(denom, (size_t)N_REL * N_NODES);

    // 2) base embedding GEMM (bf16x3 WMMA): h0 = 0.5*(ft@Wt + fi@Wi)
    uge_embed_kernel<<<mblocks, 128, 0, stream>>>(feat_text, feat_image,
                                                  Wt_hi, Wt_lo, Wi_hi, Wi_lo, h0);

    // 3) per-relation denominators
    uge_denom_kernel<<<(E + 255) / 256, 256, 0, stream>>>(dst, edge_type, edge_w, E, denom);

    // 4) per relation: zero scratch -> weighted scatter -> GEMM accumulate into out
    for (int rel = 0; rel < N_REL; ++rel) {
        uge_zero_kernel<<<2048, 256, 0, stream>>>(Aagg, NH);
        uge_scatter_kernel<<<(E + 7) / 8, 256, 0, stream>>>(h0, src, dst, edge_type, edge_w,
                                                            E, rel, Aagg);
        uge_relmm_kernel<<<mblocks, 128, 0, stream>>>(Aagg,
                                                      Wr_hi + (size_t)rel * 65536,
                                                      Wr_lo + (size_t)rel * 65536,
                                                      denom + (size_t)rel * N_NODES, out);
    }

    // 5) fused residual + LayerNorm (+ zero row 0)
    uge_ln_kernel<<<(N_NODES + 3) / 4, 128, 0, stream>>>(h0, out, ln_gamma, ln_beta);
}
